// NeuralSheafLaplacian_51642686767392
// MI455X (gfx1250) — matrix operand order repaired and checked by reading, not compiled, not run on hardware
//
#include <hip/hip_runtime.h>

typedef float  v2f   __attribute__((ext_vector_type(2)));
typedef float  v8f   __attribute__((ext_vector_type(8)));
typedef __bf16 v8bf  __attribute__((ext_vector_type(8)));
typedef __bf16 v16bf __attribute__((ext_vector_type(16)));

#define NB 65536
#define NP 16
#define NE 32
#define NF 64

// ---------------------------------------------------------------------------
// Setup kernel 1: Mmat = I - damping * (incidence^T incidence)   [16x16 f32]
// ---------------------------------------------------------------------------
__global__ void sheaf_setup_mmat(const float* __restrict__ inc,
                                 const float* __restrict__ damping,
                                 float* __restrict__ Mmat) {
  int t = threadIdx.x;            // 256 threads, 1 block
  int p = t >> 4, q = t & 15;
  float s = 0.0f;
  for (int e = 0; e < NE; ++e)
    s += inc[e * NP + p] * inc[e * NP + q];
  Mmat[t] = (p == q ? 1.0f : 0.0f) - damping[0] * s;
}

// ---------------------------------------------------------------------------
// Setup kernel 2: pre-transpose sheaf_maps into bf16 WMMA B-fragment layout.
// Fragment (e, kt, nt): lane l, half h holds
//   bf16( sheaf[e][ f = kt*32 + h + 16*(l>=16) ][ g = nt*16 + (l&15) ] )
// stored contiguously so each lane reads 32B with b128 loads.
// ---------------------------------------------------------------------------
__global__ void sheaf_setup_frag(const float* __restrict__ sheaf,
                                 __bf16* __restrict__ out) {
  int i = blockIdx.x * 256 + threadIdx.x;   // 512 blocks x 256 = 131072 halves
  int h  = i & 15;
  int l  = (i >> 4) & 31;
  int nt = (i >> 9) & 3;
  int kt = (i >> 11) & 1;
  int e  = i >> 12;
  int f = kt * 32 + h + 16 * (l >> 4);
  int g = nt * 16 + (l & 15);
  out[i] = (__bf16)sheaf[e * (NF * NF) + f * NF + g];
}

// ---------------------------------------------------------------------------
// Main kernel: one workgroup (2 waves) per tile of 16 batches.
//  Phase 1 (each wave: 8 batches): f32 WMMA 16x16x4 computes, per f-tile mt,
//    three chains sharing the same A operand (S_b^T fragments):
//      diffused^T = S_b^T x Mmat^T   -> contiguous 32B f32 global stores
//      weighted^T = S_b^T x inc^T    -> bf16 packed, single ds_store_b128
//  Phase 2 (each wave: 16 edges): bf16 WMMA 16x16x32 coboundary + norms.
// ---------------------------------------------------------------------------
__global__ __launch_bounds__(64) void sheaf_main(
    const float* __restrict__ S,        // [B,P,F] f32
    const float* __restrict__ inc,      // [E,P]   f32
    const float* __restrict__ Mmat,     // [16,16] f32 (ws)
    const __bf16* __restrict__ sheafB,  // fragment layout (ws)
    float* __restrict__ diff_out,       // [B,P,F] f32
    float* __restrict__ h1_out)         // [B]     f32
{
  __shared__ __bf16 Wlds[16 * NE * NF];   // 64 KB: weighted, bf16 [b][e][f]

  const int lane = threadIdx.x & 31;
  const int wv   = threadIdx.x >> 5;      // wave id: 0 or 1
  const int hi   = lane >> 4;             // lane half
  const int col  = lane & 15;
  const int tile = blockIdx.x;            // 4096 tiles of 16 batches

  // --- constant B fragments (f32 16x16x4 layout: K = kt*4 + 2*hi + v) ---
  // Mmat^T as B: value Mmat^T[q][p=col] = Mmat[col*16 + q]
  v2f Mfrag[4];
#pragma unroll
  for (int kt = 0; kt < 4; ++kt) {
    int q = kt * 4 + 2 * hi;
    Mfrag[kt].x = Mmat[col * 16 + q];
    Mfrag[kt].y = Mmat[col * 16 + q + 1];
  }
  // inc^T as B: value inc[e = et*16 + col][p]
  v2f Ifrag[2][4];
#pragma unroll
  for (int et = 0; et < 2; ++et)
#pragma unroll
    for (int kt = 0; kt < 4; ++kt) {
      int e = et * 16 + col;
      int p = kt * 4 + 2 * hi;
      Ifrag[et][kt].x = inc[e * NP + p];
      Ifrag[et][kt].y = inc[e * NP + p + 1];
    }

  // ============== Phase 1: diffusion^T + weighted^T (f32 WMMA) =============
  for (int i = 0; i < 8; ++i) {
    const int    bl    = wv * 8 + i;
    const size_t bglob = (size_t)tile * 16 + bl;
    const float* Sb    = S + bglob * (NP * NF);

    if (i < 7)  // prefetch next batch's S rows (4KB: 128B line per lane)
      __builtin_prefetch(Sb + NP * NF + lane * 32, 0, 3);

    // S_b^T A-fragments: value S[b][p = kt*4 + 2*hi + v][f = mt*16 + col]
    v2f sf[4][4];
#pragma unroll
    for (int mt = 0; mt < 4; ++mt)
#pragma unroll
      for (int kt = 0; kt < 4; ++kt) {
        int p = kt * 4 + 2 * hi;
        int f = mt * 16 + col;
        sf[mt][kt].x = Sb[p * NF + f];
        sf[mt][kt].y = Sb[(p + 1) * NF + f];
      }

    float* Db = diff_out + bglob * (NP * NF);
#pragma unroll
    for (int mt = 0; mt < 4; ++mt) {
      v8f dacc  = {};
      v8f wacc0 = {};
      v8f wacc1 = {};
#pragma unroll
      for (int kt = 0; kt < 4; ++kt) {
        dacc  = __builtin_amdgcn_wmma_f32_16x16x4_f32(
            false, sf[mt][kt], false, Mfrag[kt],    (short)0, dacc,  false, false);
        wacc0 = __builtin_amdgcn_wmma_f32_16x16x4_f32(
            false, sf[mt][kt], false, Ifrag[0][kt], (short)0, wacc0, false, false);
        wacc1 = __builtin_amdgcn_wmma_f32_16x16x4_f32(
            false, sf[mt][kt], false, Ifrag[1][kt], (short)0, wacc1, false, false);
      }
      const int f0 = mt * 16 + 8 * hi;   // lane's 8 contiguous f values

      // diffused^T fragment: M=f (contiguous per lane), N=p=col
      *(v8f*)(Db + col * NF + f0) = dacc;

      // weighted^T fragments: M=f (contiguous), N=e -> bf16, one b128 each
      v8bf w0, w1;
#pragma unroll
      for (int r = 0; r < 8; ++r) {
        w0[r] = (__bf16)wacc0[r];
        w1[r] = (__bf16)wacc1[r];
      }
      *(v8bf*)&Wlds[(bl * NE + col)      * NF + f0] = w0;  // e = col
      *(v8bf*)&Wlds[(bl * NE + 16 + col) * NF + f0] = w1;  // e = 16 + col
    }
  }
  __syncthreads();

  // ================= Phase 2: coboundary (bf16 WMMA) + H1 norm =============
  float h1acc[8];
#pragma unroll
  for (int r = 0; r < 8; ++r) h1acc[r] = 0.0f;

  for (int e16 = 0; e16 < 16; ++e16) {
    const int e = wv * 16 + e16;

    // A fragments (M=b=col, K=f), kt = 0,1: hoisted out of the nt loop
    v16bf Afrag[2];
    const __bf16* wp = &Wlds[(col * NE + e) * NF];
#pragma unroll
    for (int kt = 0; kt < 2; ++kt) {
      int f0 = kt * 32 + 16 * hi;
      v8bf lo = *(const v8bf*)(wp + f0);
      v8bf hh = *(const v8bf*)(wp + f0 + 8);
      Afrag[kt] = __builtin_shufflevector(lo, hh, 0, 1, 2, 3, 4, 5, 6, 7,
                                          8, 9, 10, 11, 12, 13, 14, 15);
    }

    float sq[8];
#pragma unroll
    for (int r = 0; r < 8; ++r) sq[r] = 0.0f;

#pragma unroll
    for (int nt = 0; nt < 4; ++nt) {
      v8f acc = {};
#pragma unroll
      for (int kt = 0; kt < 2; ++kt) {
        // B fragment: pre-transposed sheaf, 32B contiguous per lane
        v16bf Bv = *(const v16bf*)(
            sheafB + ((((size_t)e * 2 + kt) * 4 + nt) * 32 + lane) * 16);
        acc = __builtin_amdgcn_wmma_f32_16x16x32_bf16(
            false, Afrag[kt], false, Bv, (short)0, acc, false, false);
      }
#pragma unroll
      for (int r = 0; r < 8; ++r) sq[r] += acc[r] * acc[r];
    }
    // reduce over g within each 16-lane half (b = r + 8*hi), then sqrt
#pragma unroll
    for (int r = 0; r < 8; ++r) {
      float s = sq[r];
      s += __shfl_xor(s, 1, 32);
      s += __shfl_xor(s, 2, 32);
      s += __shfl_xor(s, 4, 32);
      s += __shfl_xor(s, 8, 32);
      h1acc[r] += sqrtf(s);
    }
  }

  __syncthreads();                       // done reading Wlds; reuse as scratch
  float* h1p = (float*)Wlds;             // [2 waves][16 batches]
  if (lane == 0) {
#pragma unroll
    for (int r = 0; r < 8; ++r) h1p[wv * 16 + r] = h1acc[r];       // b = r
  }
  if (lane == 16) {
#pragma unroll
    for (int r = 0; r < 8; ++r) h1p[wv * 16 + 8 + r] = h1acc[r];   // b = r+8
  }
  __syncthreads();
  if (threadIdx.x < 16) {
    int t = threadIdx.x;
    h1_out[(size_t)tile * 16 + t] = (h1p[t] + h1p[16 + t]) * (1.0f / 32.0f);
  }
}

// ---------------------------------------------------------------------------
extern "C" void kernel_launch(void* const* d_in, const int* in_sizes, int n_in,
                              void* d_out, int out_size, void* d_ws,
                              size_t ws_size, hipStream_t stream) {
  (void)in_sizes; (void)n_in; (void)out_size; (void)ws_size;
  const float* S       = (const float*)d_in[0];  // node_sections [B,P,F]
  const float* inc     = (const float*)d_in[1];  // incidence [E,P]
  const float* sheaf   = (const float*)d_in[2];  // sheaf_maps [E,F,F]
  const float* damping = (const float*)d_in[3];  // scalar

  float*  out    = (float*)d_out;
  float*  Mmat   = (float*)d_ws;                       // 1 KB
  __bf16* sheafB = (__bf16*)((char*)d_ws + 1024);      // 256 KB fragments

  hipLaunchKernelGGL(sheaf_setup_mmat, dim3(1), dim3(256), 0, stream,
                     inc, damping, Mmat);
  hipLaunchKernelGGL(sheaf_setup_frag, dim3(512), dim3(256), 0, stream,
                     sheaf, sheafB);
  hipLaunchKernelGGL(sheaf_main, dim3(NB / 16), dim3(64), 0, stream,
                     S, inc, Mmat, sheafB,
                     out, out + (size_t)NB * NP * NF);
}
